// ProductKeyRetrieval_42614665511415
// MI455X (gfx1250) — matrix operand order repaired and checked by reading, not compiled, not run on hardware
//
#include <hip/hip_runtime.h>

typedef __attribute__((ext_vector_type(16))) __bf16 v16bf;
typedef __attribute__((ext_vector_type(8)))  __bf16 v8bf;
typedef __attribute__((ext_vector_type(8)))  float  v8f;

#define DIMS      256
#define SUBD      128
#define SQRT_N    1024
#define TM        16      // query rows per block
#define K_FINAL   16
#define NEG_INF   (-3.0e38f)

__device__ __forceinline__ unsigned short f2bf(float f) {
  unsigned u = __builtin_bit_cast(unsigned, f);
  u += 0x7FFFu + ((u >> 16) & 1u);      // round-to-nearest-even
  return (unsigned short)(u >> 16);
}

__device__ __forceinline__ v16bf load_bfrag16(const __bf16* p) {
  v8bf lo = *(const v8bf*)(p);
  v8bf hi = *(const v8bf*)(p + 8);
  return __builtin_shufflevector(lo, hi, 0,1,2,3,4,5,6,7,
                                         8,9,10,11,12,13,14,15);
}

// ---------------- key table fp32 -> bf16 conversion ----------------
__global__ void pkr_cvt_keys(const float* __restrict__ a,
                             const float* __restrict__ b,
                             unsigned short* __restrict__ out, int n) {
  int i = blockIdx.x * blockDim.x + threadIdx.x;
  if (i < n)            out[i] = f2bf(a[i]);
  else if (i < 2 * n)   out[i] = f2bf(b[i - n]);
}

// ---------------- main kernel: LN + WMMA scores + top-k ----------------
__global__ void __launch_bounds__(256)
pkr_kernel(const float* __restrict__ query,
           const __bf16* __restrict__ keys,     // [2][1024][128] bf16
           const float* __restrict__ gamma_,
           const float* __restrict__ beta_,
           int*   __restrict__ out_idx,
           float* __restrict__ out_scr) {
  extern __shared__ char smem[];
  __bf16* q_lds = (__bf16*)smem;                       // 16 x 256 bf16   (8 KB)
  float*  sc    = (float*)(smem + 8192);               // 16 x 1024 f32   (64 KB)
  float*  tv    = (float*)(smem + 8192 + 65536);       // [2][16][16] f32 (2 KB)
  int*    ti    = (int*)  (smem + 8192 + 65536 + 2048);// [2][16][16] i32 (2 KB)

  const int tid  = threadIdx.x;
  const int lane = tid & 31;
  const int w    = tid >> 5;                 // wave id 0..7
  const size_t grow0 = (size_t)blockIdx.x * TM;

  // ---------- Phase 1: LayerNorm 16 rows, bf16 result to LDS ----------
  {
    const int half = lane >> 4;              // which row of the pair
    const int s    = lane & 15;              // 16-float chunk within row
    const int lrow = w * 2 + half;
    const float* qr = query + (grow0 + lrow) * DIMS + s * 16;
    float x[16];
    #pragma unroll
    for (int v = 0; v < 4; ++v) {
      float4 t = ((const float4*)qr)[v];
      x[v*4+0] = t.x; x[v*4+1] = t.y; x[v*4+2] = t.z; x[v*4+3] = t.w;
    }
    float sum = 0.f, sq = 0.f;
    #pragma unroll
    for (int e = 0; e < 16; ++e) { sum += x[e]; sq += x[e] * x[e]; }
    #pragma unroll
    for (int m = 8; m >= 1; m >>= 1) {       // reduce within 16-lane half
      sum += __shfl_xor(sum, m);
      sq  += __shfl_xor(sq,  m);
    }
    const float mu  = sum * (1.0f / 256.0f);
    const float var = sq * (1.0f / 256.0f) - mu * mu;
    const float rs  = rsqrtf(var + 1e-5f);
    #pragma unroll
    for (int e = 0; e < 16; ++e) {
      int k = s * 16 + e;
      float y = (x[e] - mu) * rs * gamma_[k] + beta_[k];
      q_lds[lrow * DIMS + k] = __builtin_bit_cast(__bf16, f2bf(y));
    }
  }
  __syncthreads();

  // ---------- Phase 2: per-subspace WMMA scores + top-16 ----------
  for (int sub = 0; sub < 2; ++sub) {
    // A fragments (16x32 bf16 layout): lane<16 -> K 0..15; lane>=16 -> K 16..31 (contig)
    const int arow  = lane & 15;
    const int ahalf = lane >> 4;
    const __bf16* qrow = q_lds + arow * DIMS + sub * SUBD;
    v16bf afrag[4];
    #pragma unroll
    for (int c = 0; c < 4; ++c) {
      v8bf lo = *(const v8bf*)(qrow + c * 32 + ahalf * 16);
      v8bf hi = *(const v8bf*)(qrow + c * 32 + ahalf * 16 + 8);
      afrag[c] = __builtin_shufflevector(lo, hi, 0,1,2,3,4,5,6,7,
                                                 8,9,10,11,12,13,14,15);
    }
    // B fragments (32x16 bf16): lane n -> col n, K = kbase..kbase+15 (contig)
    const int bn  = lane & 15;
    const int bkb = (lane >> 4) * 16;
    const __bf16* ksub = keys + ((size_t)sub << 17);

    // 2 tiles per iteration: 16 b128 loads in flight, 2 independent wmma chains
    #pragma unroll
    for (int i = 0; i < 4; ++i) {
      const int tileA = w + 8 * (2 * i);
      const int tileB = w + 8 * (2 * i + 1);
      const __bf16* kpA = ksub + (size_t)(tileA * 16 + bn) * SUBD + bkb;
      const __bf16* kpB = ksub + (size_t)(tileB * 16 + bn) * SUBD + bkb;
      if (i < 3) {
        __builtin_prefetch(kpA + 16 * 16 * SUBD, 0, 1);
        __builtin_prefetch(kpB + 16 * 16 * SUBD, 0, 1);
      }
      v16bf bA[4], bB[4];
      #pragma unroll
      for (int c = 0; c < 4; ++c) bA[c] = load_bfrag16(kpA + c * 32);
      #pragma unroll
      for (int c = 0; c < 4; ++c) bB[c] = load_bfrag16(kpB + c * 32);

      v8f accA = {}, accB = {};
      #pragma unroll
      for (int c = 0; c < 4; ++c) {
        accA = __builtin_amdgcn_wmma_f32_16x16x32_bf16(
            false, afrag[c], false, bA[c], (short)0, accA, false, false);
        accB = __builtin_amdgcn_wmma_f32_16x16x32_bf16(
            false, afrag[c], false, bB[c], (short)0, accB, false, false);
      }
      // D layout: VGPR r, lane l -> row = r + 8*(l>>4), col = l&15
      const int colA  = tileA * 16 + (lane & 15);
      const int colB  = tileB * 16 + (lane & 15);
      const int rbase = (lane >> 4) * 8;
      #pragma unroll
      for (int r = 0; r < 8; ++r) {
        sc[(rbase + r) * SQRT_N + colA] = accA[r];
        sc[(rbase + r) * SQRT_N + colB] = accB[r];
      }
    }
    __syncthreads();

    // top-16 of 1024 per row (only top-16 needed: joint winners have i,j < 16)
    #pragma unroll
    for (int rr = 0; rr < 2; ++rr) {
      const int row = w * 2 + rr;
      float lv[32];
      #pragma unroll
      for (int j = 0; j < 32; ++j)
        lv[j] = sc[row * SQRT_N + lane + 32 * j];
      float lbv = NEG_INF; int lbc = lane;
      #pragma unroll
      for (int j = 0; j < 32; ++j)
        if (lv[j] > lbv) { lbv = lv[j]; lbc = lane + 32 * j; }
      for (int it = 0; it < K_FINAL; ++it) {
        float bv = lbv; int bc = lbc;
        #pragma unroll
        for (int off = 16; off >= 1; off >>= 1) {
          float ov = __shfl_xor(bv, off);
          int   oc = __shfl_xor(bc, off);
          if (ov > bv || (ov == bv && oc < bc)) { bv = ov; bc = oc; }
        }
        if (lane == 0) {
          tv[sub * 256 + row * 16 + it] = bv;
          ti[sub * 256 + row * 16 + it] = bc;
        }
        if (lane == (bc & 31)) {           // owner removes winner, rescans
          lv[bc >> 5] = NEG_INF;
          lbv = NEG_INF; lbc = lane;
          #pragma unroll
          for (int j = 0; j < 32; ++j)
            if (lv[j] > lbv) { lbv = lv[j]; lbc = lane + 32 * j; }
        }
      }
    }
    __syncthreads();
  }

  // ---------- Phase 3: joint top-16 over 16x16 candidate grid ----------
  #pragma unroll
  for (int rr = 0; rr < 2; ++rr) {
    const int row = w * 2 + rr;
    const float* t0 = tv + row * 16;
    const float* t1 = tv + 256 + row * 16;
    const int*   i0 = ti + row * 16;
    const int*   i1 = ti + 256 + row * 16;
    float jv[8];
    #pragma unroll
    for (int t = 0; t < 8; ++t) {
      int c = lane * 8 + t;                 // c = i*16 + j
      jv[t] = t0[c >> 4] + t1[c & 15];
    }
    float lbv = NEG_INF; int lbc = lane * 8;
    #pragma unroll
    for (int t = 0; t < 8; ++t)
      if (jv[t] > lbv) { lbv = jv[t]; lbc = lane * 8 + t; }
    const size_t obase = (grow0 + row) * K_FINAL;
    for (int it = 0; it < K_FINAL; ++it) {
      float bv = lbv; int bc = lbc;
      #pragma unroll
      for (int off = 16; off >= 1; off >>= 1) {
        float ov = __shfl_xor(bv, off);
        int   oc = __shfl_xor(bc, off);
        if (ov > bv || (ov == bv && oc < bc)) { bv = ov; bc = oc; }
      }
      if (lane == 0) {
        int ii = bc >> 4, jj = bc & 15;
        out_idx[obase + it] = i0[ii] * SQRT_N + i1[jj];
        out_scr[obase + it] = bv;
      }
      if (lane == (bc >> 3)) {
        jv[bc & 7] = NEG_INF;
        lbv = NEG_INF; lbc = lane * 8;
        #pragma unroll
        for (int t = 0; t < 8; ++t)
          if (jv[t] > lbv) { lbv = jv[t]; lbc = lane * 8 + t; }
      }
    }
  }
}

extern "C" void kernel_launch(void* const* d_in, const int* in_sizes, int n_in,
                              void* d_out, int out_size, void* d_ws, size_t ws_size,
                              hipStream_t stream) {
  const float* query = (const float*)d_in[0];
  const float* ck    = (const float*)d_in[1];
  const float* cpk   = (const float*)d_in[2];
  const float* gm    = (const float*)d_in[3];
  const float* bt    = (const float*)d_in[4];

  const int nkeys = in_sizes[1];                 // 1024*128 = 131072
  unsigned short* keys16 = (unsigned short*)d_ws;
  const int total = 2 * nkeys;
  pkr_cvt_keys<<<(total + 255) / 256, 256, 0, stream>>>(ck, cpk, keys16, nkeys);

  const int n_rows = in_sizes[0] / DIMS;         // 65536
  int*   out_idx = (int*)d_out;
  float* out_scr = (float*)d_out + (size_t)out_size / 2;
  const size_t lds_bytes = 8192 + 65536 + 2048 + 2048;   // ~76 KB
  pkr_kernel<<<n_rows / TM, 256, lds_bytes, stream>>>(
      query, (const __bf16*)keys16, gm, bt, out_idx, out_scr);
}
